// SalienceTransformerEncoderLayer_86861418595038
// MI455X (gfx1250) — matrix operand order repaired and checked by reading, not compiled, not run on hardware
//
#include <hip/hip_runtime.h>
#include <stdint.h>

#define HHEADS 8
#define CDIM   256
#define DFFDIM 1024
#define BDIM   2
#define SDIM   13294

typedef __attribute__((ext_vector_type(16))) __bf16 v16bf;
typedef __attribute__((ext_vector_type(8)))  float  v8f;

__device__ __forceinline__ __bf16 f2bf(float f) {
    unsigned u = __builtin_bit_cast(unsigned, f);
    u += 0x7FFFu + ((u >> 16) & 1u);              // round-to-nearest-even
    unsigned short h = (unsigned short)(u >> 16);
    return __builtin_bit_cast(__bf16, h);
}

// ---------------- weight fp32[K,N] -> bf16 transposed [N,K] ----------------
__global__ void wt_transpose_kernel(const float* __restrict__ W,
                                    __bf16* __restrict__ Wt, int K, int N) {
    int i = blockIdx.x * blockDim.x + threadIdx.x;
    if (i >= K * N) return;
    int k = i / N;
    int n = i - k * N;
    Wt[(size_t)n * K + k] = f2bf(W[i]);
}

// ---------------- elementwise (a [+ b]) -> bf16 ----------------
__global__ void addcvt_kernel(const float* __restrict__ a,
                              const float* __restrict__ b,
                              __bf16* __restrict__ o, int n) {
    int i = blockIdx.x * blockDim.x + threadIdx.x;
    if (i >= n) return;
    float x = a[i] + (b ? b[i] : 0.0f);
    o[i] = f2bf(x);
}

// ---------------- bf16 WMMA GEMM: C[M,N] = A[M,K] @ Bt[N,K]^T + bias -------
// Block tile 128x128 (8 waves, wave tile 32x64 = 2x4 WMMA).  K-step 32.
// N must be a multiple of 128; OOB M rows are index-clamped on load (their
// contribution only reaches OOB output rows, which the epilogue guards).
// mode 0: fp32 out.  mode 1: relu -> bf16 out.
__global__ __launch_bounds__(256) void gemm_bf16_wmma(
    const __bf16* __restrict__ A, const __bf16* __restrict__ Bt,
    const float* __restrict__ bias, float* __restrict__ outF,
    __bf16* __restrict__ outB, int M, int N, int K, int mode)
{
    __shared__ __bf16 As[128 * 40];   // padded stride 40 (80B) -> conflict-free b128
    __shared__ __bf16 Bs[128 * 40];

    const int t     = threadIdx.x;
    const int lane  = t & 31;
    const int wid   = t >> 5;
    const int waveM = wid >> 1;       // 0..3 -> 32-row slice
    const int waveN = wid & 1;        // 0..1 -> 64-col slice
    const int blockM = blockIdx.y * 128;
    const int blockN = blockIdx.x * 128;

    const int lr0 = t >> 2;           // 0..63 (row within tile)
    const int lc  = (t & 3) * 8;      // 8-element (16B) chunk offset

    v8f acc[2][4] = {};

    union Frag { v16bf v; uint4 q[2]; };

    for (int kk = 0; kk < K; kk += 32) {
        __syncthreads();
        #pragma unroll
        for (int rr = 0; rr < 2; ++rr) {
            int arow = lr0 + rr * 64;
            // ---- A tile 128x32: clamp OOB rows (no exec-mask divergence) ----
            int grow = blockM + arow;
            grow = grow < M ? grow : M - 1;
            uint4 da = *reinterpret_cast<const uint4*>(A + (size_t)grow * K + kk + lc);
            *reinterpret_cast<uint4*>(&As[arow * 40 + lc]) = da;
            // ---- B tile 128x32: N multiple of 128, always in range ----
            uint4 db = *reinterpret_cast<const uint4*>(Bt + (size_t)(blockN + arow) * K + kk + lc);
            *reinterpret_cast<uint4*>(&Bs[arow * 40 + lc]) = db;
        }
        __syncthreads();

        // ---- fragments per ISA 16-bit layouts ----
        Frag a[2], b[4];
        const int mr = lane & 15;
        const int ka = (lane >> 4) * 8;     // A: K runs {ka..ka+7, ka+16..ka+23}
        #pragma unroll
        for (int i = 0; i < 2; ++i) {
            int r = waveM * 32 + i * 16 + mr;
            a[i].q[0] = *reinterpret_cast<const uint4*>(&As[r * 40 + ka]);
            a[i].q[1] = *reinterpret_cast<const uint4*>(&As[r * 40 + ka + 16]);
        }
        const int kb = (lane >> 4) * 8;     // B: K runs {kb..kb+7, kb+16..kb+23}
        #pragma unroll
        for (int j = 0; j < 4; ++j) {
            int r = waveN * 64 + j * 16 + mr;
            b[j].q[0] = *reinterpret_cast<const uint4*>(&Bs[r * 40 + kb]);
            b[j].q[1] = *reinterpret_cast<const uint4*>(&Bs[r * 40 + kb + 16]);
        }
        #pragma unroll
        for (int i = 0; i < 2; ++i)
            #pragma unroll
            for (int j = 0; j < 4; ++j)
                acc[i][j] = __builtin_amdgcn_wmma_f32_16x16x32_bf16(
                    false, a[i].v, false, b[j].v, (short)0, acc[i][j], false, false);
    }

    // ---- epilogue: C VGPR r -> M = r + 8*(lane>=16), N = lane&15 ----
    const int ncol = lane & 15;
    const int moff = (lane >> 4) * 8;
    #pragma unroll
    for (int i = 0; i < 2; ++i) {
        #pragma unroll
        for (int j = 0; j < 4; ++j) {
            int col = blockN + waveN * 64 + j * 16 + ncol;
            float bb = bias ? bias[col] : 0.0f;
            #pragma unroll
            for (int r = 0; r < 8; ++r) {
                int row = blockM + waveM * 32 + i * 16 + moff + r;
                if (row < M) {
                    float val = acc[i][j][r] + bb;
                    if (mode == 0)
                        outF[(size_t)row * N + col] = val;
                    else
                        outB[(size_t)row * N + col] = f2bf(fmaxf(val, 0.0f));
                }
            }
        }
    }
}

// ---------------- softmax over 16 (L*P) per (b,s,h), in place ----------------
__global__ void softmax16_kernel(float* __restrict__ aw, int rows) {
    int r = blockIdx.x * blockDim.x + threadIdx.x;
    if (r >= rows) return;
    float* p = aw + (size_t)r * 16;
    float mx = -3.4e38f;
    #pragma unroll
    for (int i = 0; i < 16; ++i) mx = fmaxf(mx, p[i]);
    float sum = 0.0f;
    #pragma unroll
    for (int i = 0; i < 16; ++i) sum += __expf(p[i] - mx);
    float inv = 1.0f / sum;
    #pragma unroll
    for (int i = 0; i < 16; ++i) p[i] = __expf(p[i] - mx) * inv;
}

// ---------------- deformable-attention bilinear sampler --------------------
// one wave per (b,s,h); lane = channel dd -> coalesced 128B corner gathers
__global__ __launch_bounds__(256) void msda_sample_kernel(
    const float* __restrict__ vf,    // [B*S, 256] value after Wv (col = h*32+dd)
    const float* __restrict__ off,   // [B*S, h, L, P, 2]
    const float* __restrict__ aw,    // [B*S, h, 16] softmaxed
    const float* __restrict__ refp,  // [B*S, 4, 2]
    __bf16* __restrict__ attn)       // [B*S, 256] bf16
{
    const int HS[4]  = {100, 50, 25, 13};
    const int WS[4]  = {100, 50, 25, 13};
    const int LSI[4] = {0, 10000, 12500, 13125};

    int w = blockIdx.x * 8 + (threadIdx.x >> 5);
    const int dd = threadIdx.x & 31;
    if (w >= BDIM * SDIM * HHEADS) return;
    const int hh = w & 7;
    const int bs = w >> 3;          // b*S + s
    const int b  = bs / SDIM;

    const float* offp = off + (size_t)bs * 256 + hh * 32;
    const float* awp  = aw  + (size_t)bs * 128 + hh * 16;

    float acc = 0.0f;
    #pragma unroll
    for (int l = 0; l < 4; ++l) {
        const float rx = refp[((size_t)bs * 4 + l) * 2 + 0];
        const float ry = refp[((size_t)bs * 4 + l) * 2 + 1];
        const int   Wl = WS[l], Hl = HS[l];
        const float fW = (float)Wl, fH = (float)Hl;
        const float* vbase = vf + ((size_t)b * SDIM + LSI[l]) * 256 + hh * 32 + dd;
        #pragma unroll
        for (int p = 0; p < 4; ++p) {
            const float ox = offp[(l * 4 + p) * 2 + 0];
            const float oy = offp[(l * 4 + p) * 2 + 1];
            const float x = (rx + ox / fW) * fW - 0.5f;
            const float y = (ry + oy / fH) * fH - 0.5f;
            const float x0f = floorf(x), y0f = floorf(y);
            const float tx = x - x0f, ty = y - y0f;
            const int x0 = (int)x0f, y0 = (int)y0f;
            const float a = awp[l * 4 + p];
            float sample = 0.0f;
            #pragma unroll
            for (int dy = 0; dy < 2; ++dy) {
                #pragma unroll
                for (int dx = 0; dx < 2; ++dx) {
                    const int ix = x0 + dx, iy = y0 + dy;
                    const bool valid = (ix >= 0) & (ix < Wl) & (iy >= 0) & (iy < Hl);
                    const int cx = ix < 0 ? 0 : (ix >= Wl ? Wl - 1 : ix);
                    const int cy = iy < 0 ? 0 : (iy >= Hl ? Hl - 1 : iy);
                    const float wgt = (dx ? tx : 1.0f - tx) * (dy ? ty : 1.0f - ty);
                    const float g = vbase[(size_t)(cy * Wl + cx) * 256];
                    sample += valid ? wgt * g : 0.0f;
                }
            }
            acc += a * sample;
        }
    }
    attn[(size_t)bs * 256 + hh * 32 + dd] = f2bf(acc);
}

// ---------------- residual add + LayerNorm (C=256), optional bf16 copy -----
__global__ __launch_bounds__(256) void ln_add_kernel(
    const float* __restrict__ x1, const float* __restrict__ x2,
    const float* __restrict__ gam, const float* __restrict__ bet,
    float* __restrict__ outF, __bf16* __restrict__ outB)
{
    __shared__ float ssum[8], ssq[8];
    const int m = blockIdx.x;
    const int t = threadIdx.x;
    const size_t idx = (size_t)m * 256 + t;
    const float x = x1[idx] + x2[idx];
    float s1 = x, s2 = x * x;
    #pragma unroll
    for (int o = 16; o > 0; o >>= 1) {
        s1 += __shfl_xor(s1, o);
        s2 += __shfl_xor(s2, o);
    }
    const int wid = t >> 5, lid = t & 31;
    if (lid == 0) { ssum[wid] = s1; ssq[wid] = s2; }
    __syncthreads();
    if (t == 0) {
        float a = 0.0f, b = 0.0f;
        #pragma unroll
        for (int i = 0; i < 8; ++i) { a += ssum[i]; b += ssq[i]; }
        ssum[0] = a; ssq[0] = b;
    }
    __syncthreads();
    const float mean = ssum[0] * (1.0f / 256.0f);
    const float var  = ssq[0] * (1.0f / 256.0f) - mean * mean;
    const float y = (x - mean) * rsqrtf(var + 1e-5f) * gam[t] + bet[t];
    outF[idx] = y;
    if (outB) outB[idx] = f2bf(y);
}

// ============================ host launch ===================================
extern "C" void kernel_launch(void* const* d_in, const int* in_sizes, int n_in,
                              void* d_out, int out_size, void* d_ws, size_t ws_size,
                              hipStream_t stream) {
    (void)in_sizes; (void)n_in; (void)out_size; (void)ws_size;
    const float* query  = (const float*)d_in[0];
    const float* qpos   = (const float*)d_in[1];
    const float* value  = (const float*)d_in[2];
    const float* refp   = (const float*)d_in[3];
    // d_in[4] score_tgt, d_in[5] foreground, d_in[6] shapes, d_in[7] lsi:
    // top-k gather/scatter is a value identity -> unused.
    const float* Wv   = (const float*)d_in[8];
    const float* bv   = (const float*)d_in[9];
    const float* Woff = (const float*)d_in[10];
    const float* boff = (const float*)d_in[11];
    const float* Waw  = (const float*)d_in[12];
    const float* baw  = (const float*)d_in[13];
    const float* Wout = (const float*)d_in[14];
    const float* bout = (const float*)d_in[15];
    const float* ln1g = (const float*)d_in[16];
    const float* ln1b = (const float*)d_in[17];
    const float* ln2g = (const float*)d_in[18];
    const float* ln2b = (const float*)d_in[19];
    const float* W1   = (const float*)d_in[20];
    const float* b1   = (const float*)d_in[21];
    const float* W2   = (const float*)d_in[22];
    const float* b2   = (const float*)d_in[23];
    float* out = (float*)d_out;

    const int ME = BDIM * SDIM;                 // 26588 rows

    char* ws = (char*)d_ws;
    size_t cur = 0;
    auto alloc = [&](size_t bytes) -> void* {
        void* p = ws + cur;
        cur += (bytes + 255) & ~(size_t)255;
        return p;
    };
    // bf16 transposed weights [N,K]
    __bf16* WvT   = (__bf16*)alloc((size_t)256 * 256 * 2);
    __bf16* WoffT = (__bf16*)alloc((size_t)256 * 256 * 2);
    __bf16* WawT  = (__bf16*)alloc((size_t)128 * 256 * 2);
    __bf16* WoutT = (__bf16*)alloc((size_t)256 * 256 * 2);
    __bf16* W1T   = (__bf16*)alloc((size_t)DFFDIM * 256 * 2);
    __bf16* W2T   = (__bf16*)alloc((size_t)256 * DFFDIM * 2);
    // activations
    __bf16* val_bf  = (__bf16*)alloc((size_t)ME * 256 * 2);
    __bf16* qpe_bf  = (__bf16*)alloc((size_t)ME * 256 * 2);
    float*  v_f32   = (float*) alloc((size_t)ME * 256 * 4);
    float*  off_f32 = (float*) alloc((size_t)ME * 256 * 4);
    float*  aw_f32  = (float*) alloc((size_t)ME * 128 * 4);
    __bf16* attn_bf = (__bf16*)alloc((size_t)ME * 256 * 2);
    float*  src2    = (float*) alloc((size_t)ME * 256 * 4);
    float*  q_f32   = (float*) alloc((size_t)ME * 256 * 4);
    __bf16* q_bf    = (__bf16*)alloc((size_t)ME * 256 * 2);
    __bf16* hid_bf  = (__bf16*)alloc((size_t)ME * DFFDIM * 2);
    float*  ffo     = src2;   // src2 is dead after LN1 -> reuse

    // 1) weights -> bf16 [N,K]
    auto wt = [&](const float* W, __bf16* Wt, int K, int N) {
        int n = K * N;
        wt_transpose_kernel<<<(n + 255) / 256, 256, 0, stream>>>(W, Wt, K, N);
    };
    wt(Wv,   WvT,   256, 256);
    wt(Woff, WoffT, 256, 256);
    wt(Waw,  WawT,  256, 128);
    wt(Wout, WoutT, 256, 256);
    wt(W1,   W1T,   256, DFFDIM);
    wt(W2,   W2T,   DFFDIM, 256);

    // 2) qpe = query + query_pos (bf16); value -> bf16
    {
        int n = ME * 256;
        addcvt_kernel<<<(n + 255) / 256, 256, 0, stream>>>(query, qpos, qpe_bf, n);
        addcvt_kernel<<<(n + 255) / 256, 256, 0, stream>>>(value, nullptr, val_bf, n);
    }

    auto gemm = [&](const __bf16* A, const __bf16* Bt, const float* bias,
                    float* oF, __bf16* oB, int M, int N, int K, int mode) {
        dim3 grid(N / 128, (M + 127) / 128);
        gemm_bf16_wmma<<<grid, 256, 0, stream>>>(A, Bt, bias, oF, oB, M, N, K, mode);
    };

    // 3) projections
    gemm(val_bf, WvT,   bv,   v_f32,   nullptr, ME, 256, 256, 0);   // value proj
    gemm(qpe_bf, WoffT, boff, off_f32, nullptr, ME, 256, 256, 0);   // sampling offsets
    gemm(qpe_bf, WawT,  baw,  aw_f32,  nullptr, ME, 128, 256, 0);   // attn weights

    // 4) softmax over L*P per (b,s,h)
    {
        int rows = ME * HHEADS;
        softmax16_kernel<<<(rows + 255) / 256, 256, 0, stream>>>(aw_f32, rows);
    }

    // 5) deformable bilinear sampling -> attn (bf16)
    {
        int waves = ME * HHEADS;
        msda_sample_kernel<<<(waves + 7) / 8, 256, 0, stream>>>(
            v_f32, off_f32, aw_f32, refp, attn_bf);
    }

    // 6) output projection
    gemm(attn_bf, WoutT, bout, src2, nullptr, ME, 256, 256, 0);

    // 7) q = LN(query + src2)  (fp32 + bf16 copies)
    ln_add_kernel<<<ME, 256, 0, stream>>>(query, src2, ln1g, ln1b, q_f32, q_bf);

    // 8) FFN: hid = relu(q@W1+b1) bf16 ; ffo = hid@W2+b2 fp32
    gemm(q_bf,   W1T, b1, nullptr, hid_bf, ME, DFFDIM, 256, 1);
    gemm(hid_bf, W2T, b2, ffo,     nullptr, ME, 256, DFFDIM, 0);

    // 9) out = LN(q + ffo)
    ln_add_kernel<<<ME, 256, 0, stream>>>(q_f32, ffo, ln2g, ln2b, out, nullptr);
}
